// SSMBranch_16389595201626
// MI455X (gfx1250) — matrix operand order repaired and checked by reading, not compiled, not run on hardware
//
#include <hip/hip_runtime.h>
#include <hip/hip_bf16.h>

// ---------- problem constants ----------
constexpr int BATCH  = 8;
constexpr int SEQ    = 4096;
constexpr int DIMX   = 512;          // embed dim
constexpr int DI     = 1024;         // d_inner
constexpr int OUTD   = 256;          // embed_dim // 2
constexpr int MTOT   = BATCH * SEQ;  // 32768 rows
constexpr int NSEG   = 16;
constexpr int SEGLEN = SEQ / NSEG;   // 256

// LDS tile geometry (halves)
constexpr int ASTR = 40;             // A row stride (32 K + pad), 16B aligned
constexpr int BSTR = 136;            // B row stride (128 N + pad), 16B aligned
constexpr int ABUF = 128 * ASTR;     // 5120 halves / buffer
constexpr int BBUF = 32 * BSTR;      // 4352 halves / buffer

// ---------- vector types ----------
typedef _Float16 v16h __attribute__((ext_vector_type(16)));
typedef _Float16 v8h  __attribute__((ext_vector_type(8)));
typedef _Float16 v4h  __attribute__((ext_vector_type(4)));
typedef float    v8f  __attribute__((ext_vector_type(8)));

__device__ __forceinline__ float softplusf(float z) {
  return z > 20.f ? z : log1pf(__expf(z));
}

// ============================================================
// Double-buffered WMMA main loop:
//   acc(128M x 128N per block) += f16(A[128xK]) @ f16(W[Kx128])
// 256 threads = 8 waves; wave w owns rows [16w,16w+16), all 128 N.
// Tile k+1 is prefetched from global into registers while WMMAs
// consume tile k from LDS (LOADcnt overlaps DScnt).
// ============================================================
__device__ __forceinline__ void wmma_mainloop(
    const float* __restrict__ Amat, int lda,
    const float* __restrict__ Wmat, int ldw,
    int K, int mBase, int nBase,
    _Float16* ldsA, _Float16* ldsB, v8f acc[8])
{
  const int tid  = threadIdx.x;
  const int wv   = tid >> 5;
  const int lane = tid & 31;
  const int hsel = lane >> 4;
  const int l16  = lane & 15;
  const int nsteps = K >> 5;

  // ---- prologue: stage tile 0 into buffer 0 ----
#pragma unroll
  for (int i = 0; i < 4; ++i) {           // A: 128 rows x 32 K
    int p = tid + i * 256, row = p >> 3, c4 = p & 7;
    float4 v = *(const float4*)(Amat + (size_t)(mBase + row) * lda + c4 * 4);
    v4h h; h[0] = (_Float16)v.x; h[1] = (_Float16)v.y;
           h[2] = (_Float16)v.z; h[3] = (_Float16)v.w;
    *(v4h*)(ldsA + row * ASTR + c4 * 4) = h;
  }
#pragma unroll
  for (int i = 0; i < 4; ++i) {           // B: 32 rows x 128 N
    int p = tid + i * 256, row = p >> 5, c4 = p & 31;
    float4 v = *(const float4*)(Wmat + (size_t)row * ldw + nBase + c4 * 4);
    v4h h; h[0] = (_Float16)v.x; h[1] = (_Float16)v.y;
           h[2] = (_Float16)v.z; h[3] = (_Float16)v.w;
    *(v4h*)(ldsB + row * BSTR + c4 * 4) = h;
  }

  for (int step = 0; step < nsteps; ++step) {
    const int buf = step & 1;
    __syncthreads();

    // ---- prefetch next K tile into registers (overlaps WMMA below) ----
    float4 pa[4], pb[4];
    const bool hasNext = (step + 1) < nsteps;
    if (hasNext) {
      const int k0n = (step + 1) << 5;
#pragma unroll
      for (int i = 0; i < 4; ++i) {
        int p = tid + i * 256, row = p >> 3, c4 = p & 7;
        pa[i] = *(const float4*)(Amat + (size_t)(mBase + row) * lda + k0n + c4 * 4);
      }
#pragma unroll
      for (int i = 0; i < 4; ++i) {
        int p = tid + i * 256, row = p >> 5, c4 = p & 31;
        pb[i] = *(const float4*)(Wmat + (size_t)(k0n + row) * ldw + nBase + c4 * 4);
      }
    }

    // ---- A fragment (16x32): lane holds row l16, hsel selects K phase ----
    const _Float16* aB = ldsA + buf * ABUF;
    const _Float16* bB = ldsB + buf * BBUF;
    v16h afrag;
    {
      const _Float16* ar = aB + (wv * 16 + l16) * ASTR;
      const int kb = hsel * 8;
      v8h lo = *(const v8h*)(ar + kb);        // K = 8h .. 8h+7
      v8h hi = *(const v8h*)(ar + 16 + kb);   // K = 16+8h .. 23+8h
#pragma unroll
      for (int q = 0; q < 8; ++q) { afrag[q] = lo[q]; afrag[8 + q] = hi[q]; }
    }
    // ---- B fragments (32x16, lane=N col l16, vec idx=K table order); 2 groups of 4 ----
    const _Float16* br = bB + l16;
#pragma unroll
    for (int g = 0; g < 2; ++g) {
      v16h bf[4];
#pragma unroll
      for (int j = 0; j < 4; ++j) {
        const _Float16* bp = br + (g * 4 + j) * 16;
#pragma unroll
        for (int q = 0; q < 8; ++q) { bf[j][q] = bp[(hsel * 8 + q) * BSTR]; bf[j][8 + q] = bp[(16 + hsel * 8 + q) * BSTR]; }
      }
#pragma unroll
      for (int j = 0; j < 4; ++j)
        acc[g * 4 + j] = __builtin_amdgcn_wmma_f32_16x16x32_f16(
            false, afrag, false, bf[j], (short)0, acc[g * 4 + j], false, false);
    }

    // ---- store prefetched tile into the other buffer ----
    if (hasNext) {
      _Float16* aO = ldsA + (buf ^ 1) * ABUF;
      _Float16* bO = ldsB + (buf ^ 1) * BBUF;
#pragma unroll
      for (int i = 0; i < 4; ++i) {
        int p = tid + i * 256, row = p >> 3, c4 = p & 7;
        v4h h; h[0] = (_Float16)pa[i].x; h[1] = (_Float16)pa[i].y;
               h[2] = (_Float16)pa[i].z; h[3] = (_Float16)pa[i].w;
        *(v4h*)(aO + row * ASTR + c4 * 4) = h;
      }
#pragma unroll
      for (int i = 0; i < 4; ++i) {
        int p = tid + i * 256, row = p >> 5, c4 = p & 31;
        v4h h; h[0] = (_Float16)pb[i].x; h[1] = (_Float16)pb[i].y;
               h[2] = (_Float16)pb[i].z; h[3] = (_Float16)pb[i].w;
        *(v4h*)(bO + row * BSTR + c4 * 4) = h;
      }
    }
  }
  __syncthreads();
}

// ============================================================
// K0: W_comb = W_in[:,1024:] @ W_res ; b_comb = b_in[1024:]@W_res + b_res
// ============================================================
__global__ void __launch_bounds__(256) k_wcomb(
    const float* __restrict__ W_in, const float* __restrict__ b_in,
    const float* __restrict__ W_res, const float* __restrict__ b_res,
    float* __restrict__ Wc, float* __restrict__ bc)
{
  const int n = threadIdx.x;
  if ((int)blockIdx.x < DIMX) {
    const int d = blockIdx.x;
    const float* wr = W_in + (size_t)d * (2 * DI) + DI;
    float acc = 0.f;
    for (int j = 0; j < DI; ++j) acc = fmaf(wr[j], W_res[j * OUTD + n], acc);
    Wc[d * OUTD + n] = acc;
  } else {
    float acc = b_res[n];
    for (int j = 0; j < DI; ++j) acc = fmaf(b_in[DI + j], W_res[j * OUTD + n], acc);
    bc[n] = acc;
  }
}

// ============================================================
// K1: xs = silu(x @ W_in[:, :1024] + b_in[:1024])     (32768 x 1024)
// ============================================================
__global__ void __launch_bounds__(256) k_gemm_in(
    const float* __restrict__ x, const float* __restrict__ W_in,
    const float* __restrict__ b_in, float* __restrict__ xs)
{
  __shared__ _Float16 lA[2 * ABUF];
  __shared__ _Float16 lB[2 * BBUF];
  v8f acc[8];
#pragma unroll
  for (int j = 0; j < 8; ++j)
#pragma unroll
    for (int r = 0; r < 8; ++r) acc[j][r] = 0.f;

  const int mBase = blockIdx.x * 128;
  const int nBase = blockIdx.y * 128;    // within [0,1024)
  wmma_mainloop(x, DIMX, W_in, 2 * DI, DIMX, mBase, nBase, lA, lB, acc);

  const int lane = threadIdx.x & 31, wv = threadIdx.x >> 5;
  const int hsel = lane >> 4, l16 = lane & 15;
#pragma unroll
  for (int j = 0; j < 8; ++j) {
    const int n = nBase + j * 16 + l16;
    const float bn = b_in[n];
#pragma unroll
    for (int r = 0; r < 8; ++r) {
      const int m = mBase + wv * 16 + hsel * 8 + r;
      const float v = acc[j][r] + bn;
      // silu via fast reciprocal (v_rcp_f32)
      xs[(size_t)m * DI + n] = v * __builtin_amdgcn_rcpf(1.f + __expf(-v));
    }
  }
}

// ============================================================
// K2: dtlow = xs @ W_dt1 + b_dt1      (32768 x 16)
// ============================================================
__global__ void __launch_bounds__(256) k_dtlow(
    const float* __restrict__ xs, const float* __restrict__ W_dt1,
    const float* __restrict__ b_dt1, float* __restrict__ dtl)
{
  const int row = blockIdx.x * 16 + (threadIdx.x >> 4);
  const int n   = threadIdx.x & 15;
  const float* xr = xs + (size_t)row * DI;
  float acc = b_dt1[n];
#pragma unroll 4
  for (int k = 0; k < DI; ++k) acc = fmaf(xr[k], W_dt1[k * 16 + n], acc);
  dtl[(size_t)row * 16 + n] = acc;
}

// ============================================================
// Scan pass 1: per (b, segment, channel):
//   P = prod(a_l), Qf/Qb = local fwd/bwd scans from zero state
//   a_l = exp(A*dt_l), b_l = Bp*u_l*dt_l,
//   dt_l = softplus(dtlow_l . W_dt2[:,d] + b_dt2[d])
// ============================================================
__global__ void __launch_bounds__(256) k_scan_pass1(
    const float* __restrict__ xs, const float* __restrict__ dtl,
    const float* __restrict__ W_dt2, const float* __restrict__ b_dt2,
    const float* __restrict__ A, const float* __restrict__ Bp,
    float* __restrict__ Pf, float* __restrict__ Qf, float* __restrict__ Qb)
{
  __shared__ float sdt[SEGLEN * 16];     // 16 KB
  const int b  = blockIdx.x >> 4;
  const int s  = blockIdx.x & 15;
  const int d  = blockIdx.y * 256 + threadIdx.x;
  const int l0 = s * SEGLEN;

  for (int i = threadIdx.x; i < SEGLEN * 16; i += 256)
    sdt[i] = dtl[((size_t)(b * SEQ + l0)) * 16 + i];
  __syncthreads();

  float w2[16];
#pragma unroll
  for (int r = 0; r < 16; ++r) w2[r] = W_dt2[r * DI + d];
  const float b2 = b_dt2[d], Ad = A[d], Bpd = Bp[d];
  const float* xp = xs + ((size_t)(b * SEQ + l0)) * DI + d;

  float P = 1.f, h = 0.f;
  for (int l = 0; l < SEGLEN; ++l) {
    __builtin_prefetch(xp + (size_t)(l + 16) * DI, 0, 1);
    const float u = xp[(size_t)l * DI];
    float z = b2;
#pragma unroll
    for (int r = 0; r < 16; ++r) z = fmaf(sdt[l * 16 + r], w2[r], z);
    const float dt = softplusf(z);
    const float a  = __expf(Ad * dt);
    P *= a;
    h = h * a + Bpd * (u * dt);
  }
  const size_t ci = ((size_t)(b * NSEG + s)) * DI + d;
  Pf[ci] = P; Qf[ci] = h;

  h = 0.f;
  for (int l = SEGLEN - 1; l >= 0; --l) {
    const float u = xp[(size_t)l * DI];
    float z = b2;
#pragma unroll
    for (int r = 0; r < 16; ++r) z = fmaf(sdt[l * 16 + r], w2[r], z);
    const float dt = softplusf(z);
    const float a  = __expf(Ad * dt);
    h = h * a + Bpd * (u * dt);
  }
  Qb[ci] = h;
}

// ============================================================
// Scan pass 2: serial carry combine over the 16 segments per channel.
// ============================================================
__global__ void __launch_bounds__(256) k_scan_pass2(
    const float* __restrict__ Pf, const float* __restrict__ Qf,
    const float* __restrict__ Qb, float* __restrict__ Hf, float* __restrict__ Hb)
{
  const int idx = blockIdx.x * 256 + threadIdx.x;   // b*DI + d
  const int b = idx >> 10, d = idx & 1023;
  float h = 0.f;
  for (int s = 0; s < NSEG; ++s) {
    const size_t ci = ((size_t)(b * NSEG + s)) * DI + d;
    Hf[ci] = h;
    h = Pf[ci] * h + Qf[ci];
  }
  h = 0.f;
  for (int s = NSEG - 1; s >= 0; --s) {
    const size_t ci = ((size_t)(b * NSEG + s)) * DI + d;
    Hb[ci] = h;
    h = Pf[ci] * h + Qb[ci];
  }
}

// ============================================================
// Scan pass 3: replay with true entry states, xy = x_ * (y_f + y_b).
// ============================================================
__global__ void __launch_bounds__(256) k_scan_pass3(
    const float* __restrict__ xs, const float* __restrict__ dtl,
    const float* __restrict__ W_dt2, const float* __restrict__ b_dt2,
    const float* __restrict__ A, const float* __restrict__ Bp,
    const float* __restrict__ Cp, const float* __restrict__ Dp,
    const float* __restrict__ Hf, const float* __restrict__ Hb,
    float* __restrict__ xy)
{
  __shared__ float sdt[SEGLEN * 16];
  const int b  = blockIdx.x >> 4;
  const int s  = blockIdx.x & 15;
  const int d  = blockIdx.y * 256 + threadIdx.x;
  const int l0 = s * SEGLEN;

  for (int i = threadIdx.x; i < SEGLEN * 16; i += 256)
    sdt[i] = dtl[((size_t)(b * SEQ + l0)) * 16 + i];
  __syncthreads();

  float w2[16];
#pragma unroll
  for (int r = 0; r < 16; ++r) w2[r] = W_dt2[r * DI + d];
  const float b2 = b_dt2[d], Ad = A[d], Bpd = Bp[d], Cpd = Cp[d], Dpd = Dp[d];
  const float* xp = xs + ((size_t)(b * SEQ + l0)) * DI + d;
  float* yp = xy + ((size_t)(b * SEQ + l0)) * DI + d;
  const size_t ci = ((size_t)(b * NSEG + s)) * DI + d;

  float h = Hf[ci];
  for (int l = 0; l < SEGLEN; ++l) {
    __builtin_prefetch(xp + (size_t)(l + 16) * DI, 0, 1);
    const float u = xp[(size_t)l * DI];
    float z = b2;
#pragma unroll
    for (int r = 0; r < 16; ++r) z = fmaf(sdt[l * 16 + r], w2[r], z);
    const float dt = softplusf(z);
    const float a  = __expf(Ad * dt);
    h = h * a + Bpd * (u * dt);
    const float y = Cpd * h + Dpd * u;
    yp[(size_t)l * DI] = u * y;
  }
  h = Hb[ci];
  for (int l = SEGLEN - 1; l >= 0; --l) {
    const float u = xp[(size_t)l * DI];
    float z = b2;
#pragma unroll
    for (int r = 0; r < 16; ++r) z = fmaf(sdt[l * 16 + r], w2[r], z);
    const float dt = softplusf(z);
    const float a  = __expf(Ad * dt);
    h = h * a + Bpd * (u * dt);
    const float y = Cpd * h + Dpd * u;
    yp[(size_t)l * DI] += u * y;   // same-thread RMW: store->load same addr ordered
  }
}

// ============================================================
// K4: out = xy @ W_out + x @ W_comb + b_out + b_comb     (32768 x 256)
// ============================================================
__global__ void __launch_bounds__(256) k_gemm_out(
    const float* __restrict__ xy, const float* __restrict__ x,
    const float* __restrict__ W_out, const float* __restrict__ b_out,
    const float* __restrict__ Wc, const float* __restrict__ bc,
    float* __restrict__ out)
{
  __shared__ _Float16 lA[2 * ABUF];
  __shared__ _Float16 lB[2 * BBUF];
  v8f acc[8];
#pragma unroll
  for (int j = 0; j < 8; ++j)
#pragma unroll
    for (int r = 0; r < 8; ++r) acc[j][r] = 0.f;

  const int mBase = blockIdx.x * 128;
  const int nBase = blockIdx.y * 128;    // within [0,256)
  wmma_mainloop(xy, DI,   W_out, OUTD, DI,   mBase, nBase, lA, lB, acc);
  wmma_mainloop(x,  DIMX, Wc,    OUTD, DIMX, mBase, nBase, lA, lB, acc);

  const int lane = threadIdx.x & 31, wv = threadIdx.x >> 5;
  const int hsel = lane >> 4, l16 = lane & 15;
#pragma unroll
  for (int j = 0; j < 8; ++j) {
    const int n = nBase + j * 16 + l16;
    const float bn = b_out[n] + bc[n];
#pragma unroll
    for (int r = 0; r < 8; ++r) {
      const int m = mBase + wv * 16 + hsel * 8 + r;
      out[(size_t)m * OUTD + n] = acc[j][r] + bn;
    }
  }
}

// ============================================================
extern "C" void kernel_launch(void* const* d_in, const int* in_sizes, int n_in,
                              void* d_out, int out_size, void* d_ws, size_t ws_size,
                              hipStream_t stream) {
  const float* x     = (const float*)d_in[0];
  const float* W_in  = (const float*)d_in[1];
  const float* b_in  = (const float*)d_in[2];
  const float* W_res = (const float*)d_in[3];
  const float* b_res = (const float*)d_in[4];
  const float* W_dt1 = (const float*)d_in[5];
  const float* b_dt1 = (const float*)d_in[6];
  const float* W_dt2 = (const float*)d_in[7];
  const float* b_dt2 = (const float*)d_in[8];
  const float* A     = (const float*)d_in[9];
  const float* Bp    = (const float*)d_in[10];
  const float* Cp    = (const float*)d_in[11];
  const float* Dp    = (const float*)d_in[12];
  const float* W_out = (const float*)d_in[13];
  const float* b_out = (const float*)d_in[14];
  float* out = (float*)d_out;

  // ---- workspace carve-out (floats) ----
  float* ws = (float*)d_ws;
  size_t o = 0;
  const size_t CARR = (size_t)BATCH * NSEG * DI;          // 131072
  float* xs  = ws + o; o += (size_t)MTOT * DI;            // 128 MB
  float* xyb = ws + o; o += (size_t)MTOT * DI;            // 128 MB
  float* dtl = ws + o; o += (size_t)MTOT * 16;            // 2 MB
  float* Pf  = ws + o; o += CARR;
  float* Qf  = ws + o; o += CARR;
  float* Qb  = ws + o; o += CARR;
  float* Hf  = ws + o; o += CARR;
  float* Hb  = ws + o; o += CARR;
  float* Wc  = ws + o; o += (size_t)DIMX * OUTD;
  float* bc  = ws + o; o += OUTD;
  (void)ws_size; (void)in_sizes; (void)n_in; (void)out_size;

  // K0: fused residual-projection weights
  k_wcomb<<<DIMX + 1, 256, 0, stream>>>(W_in, b_in, W_res, b_res, Wc, bc);
  // K1: in-projection + silu (WMMA, double-buffered)
  k_gemm_in<<<dim3(MTOT / 128, DI / 128), 256, 0, stream>>>(x, W_in, b_in, xs);
  // K2: dt low-rank projection
  k_dtlow<<<MTOT / 16, 256, 0, stream>>>(xs, W_dt1, b_dt1, dtl);
  // K3: chunked fwd+bwd selective scan (3 phases)
  k_scan_pass1<<<dim3(BATCH * NSEG, DI / 256), 256, 0, stream>>>(
      xs, dtl, W_dt2, b_dt2, A, Bp, Pf, Qf, Qb);
  k_scan_pass2<<<(BATCH * DI) / 256, 256, 0, stream>>>(Pf, Qf, Qb, Hf, Hb);
  k_scan_pass3<<<dim3(BATCH * NSEG, DI / 256), 256, 0, stream>>>(
      xs, dtl, W_dt2, b_dt2, A, Bp, Cp, Dp, Hf, Hb, xyb);
  // K4: out-projection + fused residual (WMMA, dual-K accumulate)
  k_gemm_out<<<dim3(MTOT / 128, OUTD / 128), 256, 0, stream>>>(
      xyb, x, W_out, b_out, Wc, bc, out);
}
